// MedianConv_51505247814281
// MI455X (gfx1250) — matrix operand order repaired and checked by reading, not compiled, not run on hardware
//
#include <hip/hip_runtime.h>
#include <hip/hip_bf16.h>

// ---------------- problem constants (from reference) ----------------
#define IN_C     1024
#define OUT_C    128
#define MAX_DEG  48

// GEMM staging geometry
#define KCHUNK    256                 // K columns staged per TDM transfer
#define NCHUNK    (IN_C / KCHUNK)     // 4 chunks
#define PAD_DW    4                   // TDM pad: 4 DWORDs per 256 stored
#define ROWSTRIDE (KCHUNK + PAD_DW)   // 260 words: 16B-aligned rows, 260%64=4 bank spread
#define XS_WORDS  (16 * ROWSTRIDE)    // 4160 words per buffer (16.6 KB)

typedef __attribute__((ext_vector_type(16))) _Float16     v16h;
typedef __attribute__((ext_vector_type(8)))  float        v8f;
typedef __attribute__((ext_vector_type(4)))  float        v4f;
typedef __attribute__((ext_vector_type(4)))  unsigned int u32x4;
typedef __attribute__((ext_vector_type(8)))  int          i32x8;
typedef __attribute__((ext_vector_type(4)))  int          i32x4;

#if defined(__has_builtin)
#if __has_builtin(__builtin_amdgcn_tensor_load_to_lds) && __has_builtin(__builtin_amdgcn_s_wait_tensorcnt)
#define HAVE_TDM 1
#endif
#endif

__device__ __forceinline__ unsigned lds_off(const void* p) {
    // flat pointer to LDS: low 32 bits are the LDS byte offset (aperture in high bits)
    return (unsigned)(unsigned long long)(size_t)p;
}

#ifdef HAVE_TDM
// TDM load of a 16 x KCHUNK fp32 tile (memory row stride IN_C) into LDS at
// lds_addr, inserting 4 pad DWORDs per 256 stored -> LDS row stride 260 words.
__device__ __forceinline__ void tdm_load_tile(unsigned lds_addr, const float* gptr) {
    unsigned long long ga = (unsigned long long)(size_t)gptr;
    u32x4 g0;
    g0[0] = 1u;                                   // count=1, user descriptor
    g0[1] = lds_addr;                             // LDS byte address
    g0[2] = (unsigned)ga;                         // global_addr[31:0]
    g0[3] = (unsigned)(ga >> 32) | (2u << 30);    // global_addr[56:32] | type=2 (image)
    i32x8 g1;
    g1[0] = (2 << 16)        // data_size = 4 bytes
          | (1 << 20)        // pad_enable
          | (7 << 22)        // pad_interval code 7 = every 256 DWORDs
          | (3 << 25);       // pad_amount  code 3 = 4 DWORDs
    g1[1] = (KCHUNK & 0xffff) << 16;  // tensor_dim0[15:0] @ bit48
    g1[2] = (16 << 16);               // tensor_dim0 hi = 0 | tensor_dim1 = 16
    g1[3] = (KCHUNK << 16);           // tensor_dim1 hi = 0 | tile_dim0 = 256
    g1[4] = 16;                       // tile_dim1 = 16, tile_dim2 = 0
    g1[5] = IN_C;                     // tensor_dim0_stride = 1024 elements
    g1[6] = 0;
    g1[7] = 0;
    i32x4 g2 = {0, 0, 0, 0};
    i32x4 g3 = {0, 0, 0, 0};
#if __clang_major__ >= 23
    i32x8 g4 = {0, 0, 0, 0, 0, 0, 0, 0};
    __builtin_amdgcn_tensor_load_to_lds(g0, g1, g2, g3, g4, 0);
#else
    __builtin_amdgcn_tensor_load_to_lds(g0, g1, g2, g3, 0);
#endif
}
#endif

// =====================================================================
// Kernel 0: pre-pack W (f32 [1024,128]) into f16 WMMA B-fragments.
// Fragment index (c = colTile 0..7, s = K-slab 0..31, l = lane 0..31):
//   lane l holds column n = c*16 + (l&15), K = s*32 + (l>>4)*8 + e + (e>=8 ? 8 : 0).
// One thread per (c,s,l) writes its 16 halfs contiguously (32 B).
// =====================================================================
__global__ __launch_bounds__(256)
void mc_pack_w(const float* __restrict__ W, _Float16* __restrict__ Wf16)
{
    int t = blockIdx.x * blockDim.x + threadIdx.x;     // 0..8191
    if (t >= 8 * 32 * 32) return;
    int l = t & 31;
    int s = (t >> 5) & 31;
    int c = t >> 10;
    int n = c * 16 + (l & 15);
    int kb = s * 32 + (l >> 4) * 8;
    v16h v;
    #pragma unroll
    for (int e = 0; e < 16; ++e)
        v[e] = (_Float16)W[(size_t)(kb + e + (e >= 8 ? 8 : 0)) * OUT_C + n];
    *(v16h*)(Wf16 + (size_t)t * 16) = v;
}

// =====================================================================
// Kernel 1: h = x @ W. Block = 256 threads = 8 waves; block owns one
// 16-row stripe, wave w owns column tile w. x tile is TDM double-buffered
// through LDS (exactly 64 KB of x streamed per block); B fragments come
// pre-packed in f16. 32 WMMAs per wave (8 slabs x 4 chunks).
// =====================================================================
__global__ __launch_bounds__(256)
void mc_gemm_wmma(const float* __restrict__ x,
                  const _Float16* __restrict__ Wf16,
                  float* __restrict__ h)
{
    __shared__ float xs[2][XS_WORDS];

    const int lane = threadIdx.x & 31;
    const int wave = threadIdx.x >> 5;       // 0..7 -> column tile
    const int m    = lane & 15;
    const int hid  = lane >> 4;

    const float* xtile = x + (size_t)blockIdx.x * 16 * IN_C;

#ifdef HAVE_TDM
    if (wave == 0) tdm_load_tile(lds_off(&xs[0][0]), xtile);
#endif

    v8f acc = {};
    for (int c = 0; c < NCHUNK; ++c) {
        const int buf = c & 1;
#ifdef HAVE_TDM
        if (wave == 0) {
            if (c + 1 < NCHUNK) {
                tdm_load_tile(lds_off(&xs[buf ^ 1][0]), xtile + (c + 1) * KCHUNK);
                __builtin_amdgcn_s_wait_tensorcnt(1);   // chunk c complete, c+1 in flight
            } else {
                __builtin_amdgcn_s_wait_tensorcnt(0);
            }
        }
#else
        // cooperative copy of chunk c into padded LDS buffer
        for (int i = threadIdx.x; i < 16 * (KCHUNK / 4); i += 256) {
            int r  = i / (KCHUNK / 4);
            int kk = (i % (KCHUNK / 4)) * 4;
            *(v4f*)&xs[buf][r * ROWSTRIDE + kk] =
                *(const v4f*)(xtile + (size_t)r * IN_C + c * KCHUNK + kk);
        }
#endif
        __syncthreads();                     // staged chunk visible to all waves

        #pragma unroll
        for (int s = 0; s < KCHUNK / 32; ++s) {
            // ---- A fragment from LDS (ISA 16-bit A 16x32 layout) ----
            // base is a multiple of 4 words and rows are 16B aligned -> ds_load_b128
            const int base = m * ROWSTRIDE + s * 32 + hid * 8;
            v4f a0 = *(const v4f*)&xs[buf][base];        // K = hid*8 + 0..3
            v4f a1 = *(const v4f*)&xs[buf][base + 4];    // K = hid*8 + 4..7
            v4f a2 = *(const v4f*)&xs[buf][base + 16];   // K = 16 + hid*8 + 0..3
            v4f a3 = *(const v4f*)&xs[buf][base + 20];   // K = 16 + hid*8 + 4..7
            v16h A;
            #pragma unroll
            for (int e = 0; e < 4; ++e) {
                A[e]      = (_Float16)a0[e];
                A[4 + e]  = (_Float16)a1[e];
                A[8 + e]  = (_Float16)a2[e];
                A[12 + e] = (_Float16)a3[e];
            }
            // ---- B fragment: one contiguous 32 B load per lane ----
            v16h B = *(const v16h*)(Wf16 +
                     ((size_t)(wave * 32 + c * 8 + s) * 32 + lane) * 16);

            acc = __builtin_amdgcn_wmma_f32_16x16x32_f16(
                      false, A, false, B, (short)0, acc, false, false);
        }
        __syncthreads();                     // done reading buf before it is refilled
    }

    // C/D layout: element v -> row (v + 8*hid), col (lane&15) of the tile
    float* out = h + (size_t)blockIdx.x * 16 * OUT_C + wave * 16;
    #pragma unroll
    for (int v = 0; v < 8; ++v)
        out[(size_t)(v + 8 * hid) * OUT_C + m] = acc[v];
}

// =====================================================================
// Kernel 2: per-destination slot table (grouping only; median is
// order-independent, so atomic slot order cannot change the output).
// =====================================================================
__global__ void mc_build_adj(const long long* __restrict__ ei,
                             int* __restrict__ deg,
                             int* __restrict__ slots,
                             int E, int N)
{
    int t = blockIdx.x * blockDim.x + threadIdx.x;
    if (t >= E + N) return;
    int src, dst;
    if (t < E) {
        src = (int)ei[t];        // edge_index[0][t]
        dst = (int)ei[E + t];    // edge_index[1][t]
    } else {
        src = dst = t - E;       // self loop
    }
    int s = atomicAdd(&deg[dst], 1);
    if (s < MAX_DEG) slots[dst * MAX_DEG + s] = src;
}

// =====================================================================
// Kernel 3: lower median per (node, feature). One 128-thread block per
// node; lane f owns feature f and a private LDS column (no barriers).
// h (15.4 MB) is L2-resident, so the 261 MB gather stays on-die.
// =====================================================================
__global__ __launch_bounds__(OUT_C)
void mc_median(const float* __restrict__ h,
               const int*  __restrict__ deg,
               const int*  __restrict__ slots,
               const float* __restrict__ bias,
               float* __restrict__ out)
{
    __shared__ float buf[MAX_DEG * OUT_C];   // 24.6 KB
    const int n = blockIdx.x;
    const int f = threadIdx.x;               // 0..127

    int d = deg[n];
    if (d > MAX_DEG) d = MAX_DEG;

    for (int j = 0; j < d; ++j) {
        int s = slots[n * MAX_DEG + j];
        buf[j * OUT_C + f] = h[(size_t)s * OUT_C + f];
    }

    float res = 0.0f;
    if (d > 0) {
        const int k = (d - 1) >> 1;          // lower-median rank
        for (int t = 0; t <= k; ++t) {
            float mn = buf[f];
            int   mi = 0;
            for (int j = 1; j < d; ++j) {
                float v = buf[j * OUT_C + f];
                if (v < mn) { mn = v; mi = j; }
            }
            buf[mi * OUT_C + f] = __builtin_inff();   // extract min
            res = mn;
        }
    }
    out[(size_t)n * OUT_C + f] = res + bias[f];
}

// =====================================================================
extern "C" void kernel_launch(void* const* d_in, const int* in_sizes, int n_in,
                              void* d_out, int out_size, void* d_ws, size_t ws_size,
                              hipStream_t stream)
{
    const float*     x  = (const float*)d_in[0];
    const long long* ei = (const long long*)d_in[1];
    const float*     W  = (const float*)d_in[2];
    const float*     b  = (const float*)d_in[3];
    float*           out = (float*)d_out;

    const int N = in_sizes[0] / IN_C;    // 30000
    const int E = in_sizes[1] / 2;       // 480000

    // workspace: h [N*128 f32] | deg [N i32] | slots [N*48 i32] | Wf16 [1024*128 f16]
    char*     ws    = (char*)d_ws;
    float*    h     = (float*)ws;
    size_t    off   = (size_t)N * OUT_C * sizeof(float);
    int*      deg   = (int*)(ws + off);          off += (size_t)N * sizeof(int);
    int*      slots = (int*)(ws + off);          off += (size_t)N * MAX_DEG * sizeof(int);
    _Float16* Wf16  = (_Float16*)(ws + off);

    hipMemsetAsync(deg, 0, (size_t)N * sizeof(int), stream);

    mc_pack_w<<<(8 * 32 * 32 + 255) / 256, 256, 0, stream>>>(W, Wf16);

    mc_gemm_wmma<<<N / 16, 256, 0, stream>>>(x, Wf16, h);

    int tot = E + N;
    mc_build_adj<<<(tot + 255) / 256, 256, 0, stream>>>(ei, deg, slots, E, N);

    mc_median<<<N, OUT_C, 0, stream>>>(h, deg, slots, b, out);
}
